// ForkAttention_12627203850257
// MI455X (gfx1250) — matrix operand order repaired and checked by reading, not compiled, not run on hardware
//
#include <hip/hip_runtime.h>
#include <math.h>

typedef __bf16 v16bf __attribute__((ext_vector_type(16)));
typedef __bf16 v8bf  __attribute__((ext_vector_type(8)));
typedef float  v8f   __attribute__((ext_vector_type(8)));

__device__ __forceinline__ unsigned short f2bf(float f) {
    unsigned int u = __float_as_uint(f);
    u += 0x7FFFu + ((u >> 16) & 1u);     // round to nearest even
    return (unsigned short)(u >> 16);
}

// Low 32 bits of a generic LDS pointer are the wave-relative LDS byte address
// (see ISA 10.2).
__device__ __forceinline__ unsigned lds_lo(const void* p) {
    return (unsigned)(unsigned long long)p;
}

// CDNA5 async global->LDS copy (ASYNCcnt-tracked, no VGPR round-trip).
__device__ __forceinline__ void async_copy_b128(unsigned lds_off, const void* gaddr) {
    asm volatile("global_load_async_to_lds_b128 %0, %1, off"
                 :: "v"(lds_off), "v"(gaddr)
                 : "memory");
}
__device__ __forceinline__ void wait_async() {
    asm volatile("s_wait_asynccnt 0" ::: "memory");
}

// ---------------------------------------------------------------------------
// fp32 -> bf16 (raw bits) conversion, grid-stride
// ---------------------------------------------------------------------------
__global__ __launch_bounds__(256) void f32_to_bf16_kernel(const float* __restrict__ in,
                                                          unsigned short* __restrict__ out,
                                                          long n) {
    long i = (long)blockIdx.x * blockDim.x + threadIdx.x;
    long stride = (long)gridDim.x * blockDim.x;
    for (; i < n; i += stride) out[i] = f2bf(in[i]);
}

// ---------------------------------------------------------------------------
// Tiled transpose with conversion: in fp32 [R, Cc] -> out bf16 [Cc, R], batched.
// Block 256 = 32x8, 32x32 tile staged in LDS.
// ---------------------------------------------------------------------------
__global__ __launch_bounds__(256) void transpose_f32_to_bf16(
    const float* __restrict__ in, unsigned short* __restrict__ outT,
    int R, int Cc) {
    __shared__ unsigned short tile[32][33];
    const int bz = blockIdx.z;
    const int c0 = blockIdx.x * 32;   // column tile in input (row tile in output)
    const int r0 = blockIdx.y * 32;   // row tile in input
    const float* src = in + (size_t)bz * R * Cc;
    unsigned short* dst = outT + (size_t)bz * R * Cc;
    const int tx = threadIdx.x & 31;
    const int ty = threadIdx.x >> 5;
#pragma unroll
    for (int i = 0; i < 32; i += 8)
        tile[ty + i][tx] = f2bf(src[(size_t)(r0 + ty + i) * Cc + (c0 + tx)]);
    __syncthreads();
#pragma unroll
    for (int i = 0; i < 32; i += 8)
        dst[(size_t)(c0 + ty + i) * R + (r0 + tx)] = tile[tx][ty + i];
}

// ---------------------------------------------------------------------------
// bf16 WMMA GEMM:  D[m,n] = sum_k A[m,k] * Bt[n,k]  (+ bias)
//   A : row-major [M,K] bf16;  Bt : row-major [N,K] bf16
//   BIAS_MODE: 0 = none, 1 = bias[n], 2 = bias[m]
//   OUT_BF16 : bf16 vs fp32 output
//   OUT_TR   : store D transposed as [N,M]
// Block: 256 thr = 8 waves (4 M x 2 N); block tile 128(M) x 64(N); K-step 32.
// Each wave computes a 32x32 tile = 2x2 v_wmma_f32_16x16x32_bf16.
// A/B tiles streamed into double-buffered LDS with global_load_async_to_lds.
// ---------------------------------------------------------------------------
template <int BIAS_MODE, bool OUT_BF16, bool OUT_TR>
__global__ __launch_bounds__(256) void gemm_bf16_wmma(
    const unsigned short* __restrict__ A, size_t strideA,
    const unsigned short* __restrict__ Bt, size_t strideB,
    const float* __restrict__ bias,
    void* __restrict__ Dv, size_t strideD,
    int M, int N, int K) {

    __shared__ __align__(32) unsigned short As[2][128][32];  // [m][k]
    __shared__ __align__(32) unsigned short Bs[2][64][32];   // [n][k]

    const int tid = threadIdx.x;
    const int bz  = blockIdx.z;
    const int m0  = blockIdx.y * 128;
    const int n0  = blockIdx.x * 64;

    const unsigned short* Ab = A  + (size_t)bz * strideA;
    const unsigned short* Bb = Bt + (size_t)bz * strideB;

    // ---- per-thread staging assignment (async global -> LDS) ----
    // A tile 128x32: thread -> (row = tid>>1, 16-elem chunk = tid&1), 32B via 2x b128
    // B tile  64x32: thread -> (row = tid>>2,  8-elem chunk = tid&3), 16B via 1x b128
    const int arow = tid >> 1, achk = (tid & 1) * 16;
    const int brow = tid >> 2, bchk = (tid & 3) * 8;
    const unsigned short* asrc = Ab + (size_t)(m0 + arow) * K + achk;
    const unsigned short* bsrc = Bb + (size_t)(n0 + brow) * K + bchk;
    unsigned aLds[2], bLds[2];
    aLds[0] = lds_lo(&As[0][arow][achk]);
    aLds[1] = lds_lo(&As[1][arow][achk]);
    bLds[0] = lds_lo(&Bs[0][brow][bchk]);
    bLds[1] = lds_lo(&Bs[1][brow][bchk]);

    auto stage = [&](int buf, int k0) {
        async_copy_b128(aLds[buf],      asrc + k0);
        async_copy_b128(aLds[buf] + 16, asrc + k0 + 8);
        async_copy_b128(bLds[buf],      bsrc + k0);
    };

    // ---- wave/lane mapping ----
    const int wave = tid >> 5;
    const int lane = tid & 31;
    const int half = lane >> 4;
    const int l16  = lane & 15;
    const int wm   = wave >> 1;     // 0..3 -> 32-row band
    const int wn   = wave & 1;      // 0..1 -> 32-col band

    v8f acc[2][2] = {};

    stage(0, 0);
    int cur = 0;
    for (int k0 = 0; k0 < K; k0 += 32, cur ^= 1) {
        wait_async();
        __syncthreads();
        if (k0 + 32 < K) stage(cur ^ 1, k0 + 32);

        // A 16x32 fragment: lane(l16,half) holds K = 16h..16h+15 of row l16
        v16bf af[2];
#pragma unroll
        for (int mi = 0; mi < 2; ++mi) {
            const int row = wm * 32 + mi * 16 + l16;
            v8bf lo = *reinterpret_cast<const v8bf*>(&As[cur][row][16 * half]);
            v8bf hi = *reinterpret_cast<const v8bf*>(&As[cur][row][16 * half + 8]);
            af[mi] = __builtin_shufflevector(lo, hi,
                0, 1, 2, 3, 4, 5, 6, 7, 8, 9, 10, 11, 12, 13, 14, 15);
        }
        // B 32x16 fragment: lane(l16,half) holds K = 16h..16h+15 of column l16
        v16bf bfr[2];
#pragma unroll
        for (int ni = 0; ni < 2; ++ni) {
            const int bn = wn * 32 + ni * 16 + l16;
            bfr[ni] = *reinterpret_cast<const v16bf*>(&Bs[cur][bn][16 * half]);
        }
#pragma unroll
        for (int mi = 0; mi < 2; ++mi)
#pragma unroll
            for (int ni = 0; ni < 2; ++ni)
                acc[mi][ni] = __builtin_amdgcn_wmma_f32_16x16x32_bf16(
                    false, af[mi], false, bfr[ni], (short)0, acc[mi][ni],
                    false, false);
    }

    // ---- epilogue: D layout: VGPR r, lane -> (M = r + 8*half, N = l16) ----
#pragma unroll
    for (int mi = 0; mi < 2; ++mi) {
#pragma unroll
        for (int ni = 0; ni < 2; ++ni) {
            const int gn = n0 + wn * 32 + ni * 16 + l16;
            float bias_n = 0.f;
            if (BIAS_MODE == 1) bias_n = bias[gn];
#pragma unroll
            for (int r = 0; r < 8; ++r) {
                const int gm = m0 + wm * 32 + mi * 16 + half * 8 + r;
                float v = acc[mi][ni][r];
                if (BIAS_MODE == 1) v += bias_n;
                if (BIAS_MODE == 2) v += bias[gm];
                const size_t idx = (size_t)bz * strideD +
                    (OUT_TR ? (size_t)gn * M + gm : (size_t)gm * N + gn);
                if (OUT_BF16)
                    reinterpret_cast<unsigned short*>(Dv)[idx] = f2bf(v);
                else
                    reinterpret_cast<float*>(Dv)[idx] = v;
            }
        }
    }
}

// ---------------------------------------------------------------------------
// Row softmax: 2048 fp32 cols -> bf16, one block (256 thr) per row
// ---------------------------------------------------------------------------
__global__ __launch_bounds__(256) void softmax_rows_bf16(const float* __restrict__ in,
                                                         unsigned short* __restrict__ out,
                                                         float scale) {
    const long row = blockIdx.x;
    const float* rin = in + row * 2048L;
    unsigned short* rout = out + row * 2048L;
    const int tid = threadIdx.x;
    __shared__ float red[8];

    float vals[8];
    float lmax = -3.4e38f;
#pragma unroll
    for (int i = 0; i < 8; ++i) {
        float v = rin[tid + i * 256] * scale;
        vals[i] = v;
        lmax = fmaxf(lmax, v);
    }
    for (int off = 16; off > 0; off >>= 1)
        lmax = fmaxf(lmax, __shfl_xor(lmax, off));
    if ((tid & 31) == 0) red[tid >> 5] = lmax;
    __syncthreads();
    float rowmax = red[0];
#pragma unroll
    for (int w = 1; w < 8; ++w) rowmax = fmaxf(rowmax, red[w]);
    __syncthreads();

    float lsum = 0.f;
#pragma unroll
    for (int i = 0; i < 8; ++i) {
        float e = __expf(vals[i] - rowmax);
        vals[i] = e;
        lsum += e;
    }
    for (int off = 16; off > 0; off >>= 1)
        lsum += __shfl_xor(lsum, off);
    if ((tid & 31) == 0) red[tid >> 5] = lsum;
    __syncthreads();
    float rsum = 0.f;
#pragma unroll
    for (int w = 0; w < 8; ++w) rsum += red[w];
    const float inv = 1.0f / rsum;
#pragma unroll
    for (int i = 0; i < 8; ++i)
        rout[tid + i * 256] = f2bf(vals[i] * inv);
}

// ---------------------------------------------------------------------------
// Host-side orchestration
// ---------------------------------------------------------------------------
extern "C" void kernel_launch(void* const* d_in, const int* in_sizes, int n_in,
                              void* d_out, int out_size, void* d_ws, size_t ws_size,
                              hipStream_t stream) {
    const float* x        = (const float*)d_in[0];
    const float* fc_w     = (const float*)d_in[1];
    const float* fc_b     = (const float*)d_in[2];
    const float* alt_fc_w = (const float*)d_in[3];
    const float* alt_fc_b = (const float*)d_in[4];
    const float* vfc_w    = (const float*)d_in[5];
    const float* vfc_b    = (const float*)d_in[6];
    float* out = (float*)d_out;

    constexpr int  Bn = 8, C = 2048, E = 1024;
    constexpr size_t CE = (size_t)C * E;        // 2M elems
    constexpr size_t CC = (size_t)C * C;        // 4M elems
    constexpr size_t EE = (size_t)E * E;        // 1M elems

    char* ws = (char*)d_ws;
    size_t off = 0;
    auto alloc = [&](size_t bytes) -> char* {
        char* p = ws + off;
        off += (bytes + 255) & ~(size_t)255;
        return p;
    };
    unsigned short* xb    = (unsigned short*)alloc(Bn * CE * 2);  // x bf16 [b][c][e]
    unsigned short* xTb   = (unsigned short*)alloc(Bn * CE * 2);  // x^T bf16 [b][e][c]
    unsigned short* fcwb  = (unsigned short*)alloc(EE * 2);
    unsigned short* altwb = (unsigned short*)alloc(CC * 2);
    unsigned short* vfcwb = (unsigned short*)alloc(EE * 2);
    unsigned short* yb    = (unsigned short*)alloc(Bn * CE * 2);  // y bf16 [b][c][f]
    unsigned short* zb    = (unsigned short*)alloc(Bn * CE * 2);  // yx^T bf16 [b][k][e]
    unsigned short* vxTb  = (unsigned short*)alloc(Bn * CE * 2);  // vx^T bf16 [b][f][k]
    float*          attnf = (float*)alloc(Bn * CC * 4);           // scores fp32
    unsigned short* attnb = (unsigned short*)alloc(Bn * CC * 2);  // softmax bf16

    // 1) convert to bf16 (+ transposed copy of x for the context-dim GEMM)
    f32_to_bf16_kernel<<<dim3(2048), dim3(256), 0, stream>>>(x, xb, (long)(Bn * CE));
    f32_to_bf16_kernel<<<dim3(512),  dim3(256), 0, stream>>>(fc_w, fcwb, (long)EE);
    f32_to_bf16_kernel<<<dim3(2048), dim3(256), 0, stream>>>(alt_fc_w, altwb, (long)CC);
    f32_to_bf16_kernel<<<dim3(512),  dim3(256), 0, stream>>>(vfc_w, vfcwb, (long)EE);
    transpose_f32_to_bf16<<<dim3(E / 32, C / 32, Bn), dim3(256), 0, stream>>>(
        x, xTb, C, E);

    // 2) y[b,c,f] = x[b,c,:] . fc_w[f,:] + fc_b[f]
    dim3 gY(E / 64, C / 128, Bn);
    gemm_bf16_wmma<1, true, false><<<gY, 256, 0, stream>>>(
        xb, CE, fcwb, 0, fc_b, yb, CE, C, E, E);

    // 3) z[b,k,e] = yx[b,e,k] = alt_fc_w[k,:] . x^T[b,e,:] + alt_fc_b[k]
    //    A = alt_fc_w [M=k,K=c] (shared), Bt = x^T [N=e, K=c], bias over M
    gemm_bf16_wmma<2, true, false><<<gY, 256, 0, stream>>>(
        altwb, 0, xTb, CE, alt_fc_b, zb, CE, C, E, C);

    // 4) vx[b,k,f] = x[b,k,:] . vfc_w[f,:] + vfc_b[f], stored transposed [b][f][k]
    gemm_bf16_wmma<1, true, true><<<gY, 256, 0, stream>>>(
        xb, CE, vfcwb, 0, vfc_b, vxTb, CE, C, E, E);

    // 5) attn[b,c,k] = y[b,c,:] . z[b,k,:]   (fp32 out)
    dim3 gS(C / 64, C / 128, Bn);
    gemm_bf16_wmma<0, false, false><<<gS, 256, 0, stream>>>(
        yb, CE, zb, CE, nullptr, attnf, CC, C, C, E);

    // 6) softmax over k (scale 1/sqrt(2048)) -> bf16
    softmax_rows_bf16<<<dim3(Bn * C), dim3(256), 0, stream>>>(
        attnf, attnb, 1.0f / sqrtf(2048.0f));

    // 7) out[b,c,e] = attn[b,c,:] . vx^T[b,e,:]   (fp32 -> d_out)
    gemm_bf16_wmma<0, false, false><<<gY, 256, 0, stream>>>(
        attnb, CC, vxTb, CE, nullptr, out, CE, C, E, C);

    (void)in_sizes; (void)n_in; (void)out_size; (void)ws_size;
}
